// CNF_60266981097661
// MI455X (gfx1250) — matrix operand order repaired and checked by reading, not compiled, not run on hardware
//
#include <hip/hip_runtime.h>
#include <hip/hip_bf16.h>

// ---------------------------------------------------------------------------
// CNF (FFJORD-style) dopri5 integrator, fully fused, WMMA f32<-f16 on gfx1250.
// Each wave32 owns 16 batch rows for the entire 8-step integration.
// Weights are DMA'd into LDS via the Tensor Data Mover, then converted to
// f16 images in both major orders so every WMMA B-fragment is two
// 16-byte ds_load_b128 reads. tanh uses the native gfx1250 v_tanh_f32 TRANS op.
// ---------------------------------------------------------------------------

typedef _Float16 half8  __attribute__((ext_vector_type(8)));
typedef _Float16 v16h   __attribute__((ext_vector_type(16)));
typedef float    v8f    __attribute__((ext_vector_type(8)));
typedef unsigned uint4v __attribute__((ext_vector_type(4)));
typedef int      int8v  __attribute__((ext_vector_type(8)));

#define DD   64     // feature dim
#define HH   256    // hidden dim
#define NWAVE 4     // waves per workgroup
#define ROWS_PER_WG (NWAVE * 16)

// dopri5 tableau
__constant__ float Atab[6][5] = {
    {0.f, 0.f, 0.f, 0.f, 0.f},
    {0.2f, 0.f, 0.f, 0.f, 0.f},
    {0.075f, 0.225f, 0.f, 0.f, 0.f},
    {44.0f/45.0f, -56.0f/15.0f, 32.0f/9.0f, 0.f, 0.f},
    {19372.0f/6561.0f, -25360.0f/2187.0f, 64448.0f/6561.0f, -212.0f/729.0f, 0.f},
    {9017.0f/3168.0f, -355.0f/33.0f, 46732.0f/5247.0f, 49.0f/176.0f, -5103.0f/18656.0f},
};
__constant__ float Ctab[6] = {0.f, 0.2f, 0.3f, 0.8f, 8.0f/9.0f, 1.0f};
__constant__ float Btab[6] = {35.0f/384.0f, 0.f, 500.0f/1113.0f, 125.0f/192.0f,
                              -2187.0f/6784.0f, 11.0f/84.0f};

// LDS byte budget
#define SZ_WTS   (4 * HH * DD * 2)               // 4 f16 weight images (128 KB)
#define SZ_VEC   (HH*4 + HH*4 + DD*4)            // b1, t-row of W1, b2 (f32)
#define PW_KB    (6 * 16 * DD * 2)               // 6 k-stages, f16
#define PW_ZA    (16 * DD * 2)                   // z staging (A-layout source)
#define PW_HB    (16 * HH * 2)                   // h staging, f16
#define PW_GA    (16 * 32 * 2)                   // g_u 16x32 A staging
#define PW_DIV   (6 * 16 * 4)                    // per-stage divergences
#define PW_BYTES (PW_KB + PW_ZA + PW_HB + PW_GA + PW_DIV)
#define SMEM_BYTES (SZ_WTS + SZ_VEC + NWAVE * PW_BYTES)

// Native gfx1250 tanh (TRANS op, co-executes with the WMMA/XDL pipe)
__device__ __forceinline__ float fast_tanh(float x) {
#if __has_builtin(__builtin_amdgcn_tanhf)
    return __builtin_amdgcn_tanhf(x);
#elif __has_builtin(__builtin_amdgcn_tanh_f32)
    return __builtin_amdgcn_tanh_f32(x);
#else
    // tanh(x) = 1 - 2/(exp2(2x*log2e)+1)  : v_exp_f32 + v_rcp_f32
    float e2 = __builtin_amdgcn_exp2f(x * 2.8853900817779268f);
    return 1.0f - 2.0f * __builtin_amdgcn_rcpf(e2 + 1.0f);
#endif
}

// Tensor Data Mover: 2-D tile load Global -> LDS (D# per ISA 08_async_tensor).
// dim0 = innermost length (elements), dim1 = outer length, 4-byte elements.
__device__ __forceinline__ void tdm_load_2d(const void* gptr, unsigned ldsOff,
                                            unsigned dim0, unsigned dim1,
                                            unsigned stride0) {
    unsigned long long ga = (unsigned long long)gptr;
    uint4v g0;
    g0[0] = 1u;                                          // count=1, user D#
    g0[1] = ldsOff;                                      // lds_addr
    g0[2] = (unsigned)(ga & 0xffffffffu);                // global_addr[31:0]
    g0[3] = (unsigned)((ga >> 32) & 0x01ffffffu)         // global_addr[56:32]
          | (2u << 30);                                  // type = 2 ("image")
    int8v g1;
    g1[0] = (int)(2u << 16);                             // data_size = 4 bytes
    g1[1] = (int)((dim0 & 0xffffu) << 16);               // tensor_dim0[15:0]
    g1[2] = (int)(((dim0 >> 16) & 0xffffu)               // tensor_dim0[31:16]
          | ((dim1 & 0xffffu) << 16));                   // tensor_dim1[15:0]
    g1[3] = (int)(((dim1 >> 16) & 0xffffu)               // tensor_dim1[31:16]
          | ((dim0 & 0xffffu) << 16));                   // tile_dim0 = dim0
    g1[4] = (int)(dim1 & 0xffffu);                       // tile_dim1 = dim1
    g1[5] = (int)stride0;                                // tensor_dim0_stride lo
    g1[6] = 0;
    g1[7] = 0;
    uint4v g2 = {0u, 0u, 0u, 0u};                        // dims 2..4 unused
    uint4v g3 = {0u, 0u, 0u, 0u};
    asm volatile("tensor_load_to_lds %0, %1, %2, %3"
                 :: "s"(g0), "s"(g1), "s"(g2), "s"(g3)
                 : "memory");
}

__device__ __forceinline__ void wait_tensorcnt0() {
#if __has_builtin(__builtin_amdgcn_s_wait_tensorcnt)
    __builtin_amdgcn_s_wait_tensorcnt(0);
#else
    asm volatile("s_wait_tensorcnt 0x0" ::: "memory");
#endif
}

__device__ __forceinline__ v8f wmma_f16(v16h a, v16h b, v8f c) {
    return __builtin_amdgcn_wmma_f32_16x16x32_f16(false, a, false, b,
                                                  (short)0, c, false, false);
}

// A-fragment (16x32 f16) from a row-major [16][ldk] f16 buffer.
// ISA 7.12.2: lane row = lane%16; K-chunks [kb..kb+7] and [kb+16..kb+23],
// kb = 32*kk + (lane>=16 ? 8 : 0). Two contiguous 16-byte ds_load_b128.
__device__ __forceinline__ v16h ldA(const _Float16* buf, int ldk, int kk, int lane) {
    int row = lane & 15;
    int kb  = kk * 32 + ((lane & 16) ? 8 : 0);
    half8 lo = *(const half8*)(buf + row * ldk + kb);
    half8 hi = *(const half8*)(buf + row * ldk + kb + 16);
    return __builtin_shufflevector(lo, hi, 0,1,2,3,4,5,6,7,8,9,10,11,12,13,14,15);
}

// B-fragment (32x16 f16) from an [N][K] f16 buffer (K contiguous).
// lane col N = ntile*16 + lane%16; K-chunks [kb..kb+7] and [kb+16..kb+23], kb = 32*kk + (lane>=16 ? 8 : 0).
__device__ __forceinline__ v16h ldB(const _Float16* buf, int ldk, int ntile, int kk, int lane) {
    int ncol = ntile * 16 + (lane & 15);
    int kb   = kk * 32 + ((lane & 16) ? 8 : 0);
    const half8* p = (const half8*)(buf + ncol * ldk + kb);
    half8 lo = p[0];
    half8 hi = p[2];
    return __builtin_shufflevector(lo, hi, 0,1,2,3,4,5,6,7,8,9,10,11,12,13,14,15);
}

__global__ __launch_bounds__(NWAVE * 32, 1)
void cnf_dopri5_wmma(const float* __restrict__ y,  const float* __restrict__ e,
                     const float* __restrict__ W1, const float* __restrict__ b1,
                     const float* __restrict__ W2, const float* __restrict__ b2,
                     float* __restrict__ out, int Bn) {
    extern __shared__ char smem[];
    // Weight images (f16), laid out [N][K] for each of the 4 matmuls:
    _Float16* w1aT = (_Float16*)smem;              // [HH][DD]  : u   = z @ W1a
    _Float16* w1aR = w1aT + HH * DD;               // [DD][HH]  : ef  = g @ W1aT
    _Float16* w2T  = w1aR + DD * HH;               // [DD][HH]  : dz  = h @ W2
    _Float16* w2R  = w2T  + DD * HH;               // [HH][DD]  : gh  = e @ W2T
    float* b1s  = (float*)(w2R + HH * DD);         // [HH]
    float* w1ts = b1s + HH;                        // [HH] (t-row of W1)
    float* b2s  = w1ts + HH;                       // [DD]
    char*  pwBase = (char*)(b2s + DD);

    const int tid  = threadIdx.x;
    const int lane = tid & 31;
    const int wave = tid >> 5;
    const int col  = lane & 15;            // N (C-layout) / M (A-layout)
    const int hiO  = (lane & 16) ? 8 : 0;  // C-layout M offset

    _Float16* kB  = (_Float16*)(pwBase + wave * PW_BYTES);   // [6][16][DD]
    _Float16* zA  = kB + 6 * 16 * DD;                        // [16][DD]
    _Float16* hB  = zA + 16 * DD;                            // [16][HH]
    _Float16* gA  = hB + 16 * HH;                            // [16][32]
    float*    dvB = (float*)(gA + 16 * 32);                  // [6][16]

    __builtin_prefetch(y, 0, 0);
    __builtin_prefetch(e, 0, 0);

    // ---- TDM: DMA raw f32 weights into LDS staging (aliases per-wave bufs,
    //      which are not live until after weight-image construction) ----
    float* stage = (float*)pwBase;                 // >= 95 KB available
    unsigned ldsBase = (unsigned)__builtin_amdgcn_groupstaticsize();
    unsigned stageOff = ldsBase + (unsigned)((char*)stage - smem);

    if (wave == 0) {
        tdm_load_2d(W1, stageOff, HH, DD + 1, HH); // 65 x 256 f32 (66.5 KB)
        wait_tensorcnt0();
    }
    __syncthreads();
    // ---- build f16 W1 images (both major orders) + bias vectors from LDS ----
    for (int idx = tid; idx < DD * HH; idx += blockDim.x) {
        int k = idx / HH, n = idx % HH;            // k: input dim, n: hidden
        float w = stage[k * HH + n];
        w1aR[k * HH + n] = (_Float16)w;
        w1aT[n * DD + k] = (_Float16)w;
    }
    for (int n = tid; n < HH; n += blockDim.x) {
        b1s[n]  = b1[n];
        w1ts[n] = stage[DD * HH + n];              // row 64 of W1 (time column)
    }
    for (int d = tid; d < DD; d += blockDim.x) b2s[d] = b2[d];
    __syncthreads();

    if (wave == 0) {
        tdm_load_2d(W2, stageOff, DD, HH, DD);     // 256 x 64 f32 (64 KB)
        wait_tensorcnt0();
    }
    __syncthreads();
    for (int idx = tid; idx < HH * DD; idx += blockDim.x) {
        int hr = idx / DD, dc = idx % DD;          // hr: hidden, dc: feature
        float w = stage[hr * DD + dc];
        w2R[hr * DD + dc] = (_Float16)w;
        w2T[dc * HH + hr] = (_Float16)w;
    }
    __syncthreads();

    // ---- per-wave persistent state: 16 rows, C-layout f32 ----
    const int rowBase = (blockIdx.x * NWAVE + wave) * 16;
    v8f z0[4];
    v8f eC[4];
    #pragma unroll
    for (int t4 = 0; t4 < 4; ++t4) {
        #pragma unroll
        for (int r = 0; r < 8; ++r) {
            int g = (rowBase + r + hiO) * DD + t4 * 16 + col;
            z0[t4][r] = y[g];
            eC[t4][r] = e[g];
        }
    }
    // e in A-layout f16 (two 16x32 K-fragments over K=64)
    v16h eA[2];
    {
        int arow = rowBase + col;
        int base = (lane & 16) ? 8 : 0;
        #pragma unroll
        for (int kk = 0; kk < 2; ++kk) {
            v16h v;
            #pragma unroll
            for (int j = 0; j < 8; ++j) {
                v[j]     = (_Float16)e[arow * DD + kk * 32 + base + j];
                v[8 + j] = (_Float16)e[arow * DD + kk * 32 + base + 16 + j];
            }
            eA[kk] = v;
        }
    }
    float lpReg = 0.0f;
    const float dtv = 1.0f / 8.0f;

    for (int step = 0; step < 8; ++step) {
        float t0 = (float)step * dtv;
        for (int st = 0; st < 6; ++st) {
            float ts = t0 + Ctab[st] * dtv;

            // ---- yi = z0 + dt * sum_j a_ij k_j  ->  f16 A-staging ----
            #pragma unroll
            for (int t4 = 0; t4 < 4; ++t4) {
                v8f yi = z0[t4];
                for (int j = 0; j < st; ++j) {
                    float aij = Atab[st][j] * dtv;
                    #pragma unroll
                    for (int r = 0; r < 8; ++r)
                        yi[r] += aij * (float)kB[j * 16 * DD + (r + hiO) * DD + t4 * 16 + col];
                }
                #pragma unroll
                for (int r = 0; r < 8; ++r)
                    zA[(r + hiO) * DD + t4 * 16 + col] = (_Float16)yi[r];
            }
            __syncthreads();

            // ---- forward matmul 1: u = zcat @ W1, h = tanh(u + b1 + t*W1_t) ----
            v16h a0 = ldA(zA, DD, 0, lane);
            v16h a1 = ldA(zA, DD, 1, lane);
            #pragma unroll
            for (int n = 0; n < 16; ++n) {
                v8f acc = {};
                acc = wmma_f16(a0, ldB(w1aT, DD, n, 0, lane), acc);
                acc = wmma_f16(a1, ldB(w1aT, DD, n, 1, lane), acc);
                float bias = b1s[n * 16 + col] + ts * w1ts[n * 16 + col];
                #pragma unroll
                for (int r = 0; r < 8; ++r) {
                    float hv = fast_tanh(acc[r] + bias);
                    hB[(r + hiO) * HH + n * 16 + col] = (_Float16)hv;
                }
            }
            __syncthreads();

            // ---- forward matmul 2: dz = h @ W2 + b2  ->  k-stage store ----
            #pragma unroll
            for (int n = 0; n < 4; ++n) {
                v8f acc = {};
                #pragma unroll
                for (int kk = 0; kk < 8; ++kk)
                    acc = wmma_f16(ldA(hB, HH, kk, lane), ldB(w2T, HH, n, kk, lane), acc);
                float bb = b2s[n * 16 + col];
                #pragma unroll
                for (int r = 0; r < 8; ++r)
                    kB[st * 16 * DD + (r + hiO) * DD + n * 16 + col] = (_Float16)(acc[r] + bb);
            }

            // ---- backward (VJP): gh = e @ W2^T ; gu = gh*(1-h^2) ; ef = gu @ W1a^T ----
            v8f ef[4] = {};
            #pragma unroll
            for (int p = 0; p < 8; ++p) {       // K-chunks of 32 over H=256
                #pragma unroll
                for (int sub = 0; sub < 2; ++sub) {
                    int n3 = p * 2 + sub;       // g_h N-tile 0..15
                    v8f g = {};
                    g = wmma_f16(eA[0], ldB(w2R, DD, n3, 0, lane), g);
                    g = wmma_f16(eA[1], ldB(w2R, DD, n3, 1, lane), g);
                    #pragma unroll
                    for (int r = 0; r < 8; ++r) {
                        float hv = (float)hB[(r + hiO) * HH + n3 * 16 + col];
                        gA[(r + hiO) * 32 + sub * 16 + col] =
                            (_Float16)(g[r] * (1.0f - hv * hv));
                    }
                }
                __syncthreads();
                v16h ag = ldA(gA, 32, 0, lane);
                #pragma unroll
                for (int n = 0; n < 4; ++n)
                    ef[n] = wmma_f16(ag, ldB(w1aR, HH, n, p, lane), ef[n]);
                __syncthreads();
            }

            // ---- div(M) = sum_N ef*e  (wave32 half-group reduction) ----
            #pragma unroll
            for (int r = 0; r < 8; ++r) {
                float pr = 0.0f;
                #pragma unroll
                for (int t4 = 0; t4 < 4; ++t4) pr += ef[t4][r] * eC[t4][r];
                pr += __shfl_xor(pr, 1, 32);
                pr += __shfl_xor(pr, 2, 32);
                pr += __shfl_xor(pr, 4, 32);
                pr += __shfl_xor(pr, 8, 32);
                if (col == 0) dvB[st * 16 + r + hiO] = pr;
            }
        }
        __syncthreads();

        // ---- combine stages: z0 += dt * sum_j b_j k_j ; lp -= dt * sum b_j div_j ----
        #pragma unroll
        for (int t4 = 0; t4 < 4; ++t4) {
            #pragma unroll
            for (int j = 0; j < 6; ++j) {
                float c = Btab[j] * dtv;
                #pragma unroll
                for (int r = 0; r < 8; ++r)
                    z0[t4][r] += c * (float)kB[j * 16 * DD + (r + hiO) * DD + t4 * 16 + col];
            }
        }
        #pragma unroll
        for (int j = 0; j < 6; ++j)
            lpReg -= dtv * Btab[j] * dvB[j * 16 + col];
        __syncthreads();
    }

    // ---- outputs: z, then log_px = log_pz(z) - dlogp ----
    #pragma unroll
    for (int t4 = 0; t4 < 4; ++t4)
        #pragma unroll
        for (int r = 0; r < 8; ++r)
            out[(rowBase + r + hiO) * DD + t4 * 16 + col] = z0[t4][r];

    #pragma unroll
    for (int r = 0; r < 8; ++r) {
        float s2 = 0.0f;
        #pragma unroll
        for (int t4 = 0; t4 < 4; ++t4) s2 += z0[t4][r] * z0[t4][r];
        s2 += __shfl_xor(s2, 1, 32);
        s2 += __shfl_xor(s2, 2, 32);
        s2 += __shfl_xor(s2, 4, 32);
        s2 += __shfl_xor(s2, 8, 32);
        if (col == 0) dvB[r + hiO] = s2;
    }
    __syncthreads();
    {
        const float LN2PI = 1.8378770664093453f;
        float s2 = dvB[col];
        float logpz = -0.5f * (float)DD * LN2PI - 0.5f * s2;
        if (lane < 16) out[Bn * DD + rowBase + col] = logpz - lpReg;
    }
}

extern "C" void kernel_launch(void* const* d_in, const int* in_sizes, int n_in,
                              void* d_out, int out_size, void* d_ws, size_t ws_size,
                              hipStream_t stream) {
    (void)n_in; (void)out_size; (void)d_ws; (void)ws_size;
    const float* y  = (const float*)d_in[0];
    const float* e  = (const float*)d_in[1];
    const float* W1 = (const float*)d_in[2];
    const float* b1 = (const float*)d_in[3];
    const float* W2 = (const float*)d_in[4];
    const float* b2 = (const float*)d_in[5];
    float* out = (float*)d_out;

    int Bn = in_sizes[0] / DD;                     // 32768
    dim3 grid(Bn / ROWS_PER_WG);                   // 512 workgroups
    dim3 block(NWAVE * 32);                        // 4 wave32s

    hipFuncSetAttribute((const void*)cnf_dopri5_wmma,
                        hipFuncAttributeMaxDynamicSharedMemorySize,
                        (int)SMEM_BYTES);
    cnf_dopri5_wmma<<<grid, block, SMEM_BYTES, stream>>>(y, e, W1, b1, W2, b2, out, Bn);
}
